// SourceGNN_59287728554035
// MI455X (gfx1250) — compile-verified, matrix operand order checked by reading, not run on hardware
//
#include <hip/hip_runtime.h>
#include <hip/hip_bf16.h>

#define NS 100000   // sources (nodes)
#define NM 306      // sensors
#define NT 256      // times
#define NF 8        // vertex features
#define NH 128      // hidden
#define NL 3        // layers
#define NE 600000   // edges
#define INDIM (NT + NF)
#define REG2 1e-6f  // reg^2

typedef __attribute__((ext_vector_type(16))) __bf16 v16bf;
typedef __attribute__((ext_vector_type(8)))  float  v8f;
typedef __attribute__((ext_vector_type(4)))  unsigned u32x4;
typedef __attribute__((ext_vector_type(8)))  unsigned u32x8;

// K index inside a 32-wide chunk for fragment element i (ISA 16-bit A/B layout)
__device__ __forceinline__ int frag_k(int i, int half) {
    return (i & 7) + ((i >> 3) << 4) + (half << 3);
}

// ---- Tensor Data Mover: 2D tile (tile0 x tile1, 4-byte elems) -> LDS ----
// LDS padding: 1 DWORD after every 16 DWORDs -> row stride 17 floats.
// rem0/rem1 = remaining tensor extent from tile start (OOB rows zero-fill).
__device__ __forceinline__ void tdm_load_2d(unsigned lds_off, const void* gaddr,
                                            unsigned rem0, unsigned rem1,
                                            unsigned tile0, unsigned tile1,
                                            unsigned stride0) {
    unsigned long long ga = (unsigned long long)gaddr;
    u32x4 g0;
    g0[0] = 1u;                                   // count=1, user mode
    g0[1] = lds_off;                              // LDS byte address
    g0[2] = (unsigned)ga;                         // global_addr[31:0] (bit 64+)
    g0[3] = (unsigned)(ga >> 32) | (2u << 30);    // addr[56:32], type=2 (image)
    u32x8 g1;
    g1[0] = (2u << 16)       // data_size = 4 bytes
          | (1u << 20)       // pad_enable
          | (3u << 22);      // pad_interval code 3 = 16 DWORDs (pad_amount=1 DW)
    g1[1] = (rem0 & 0xFFFFu) << 16;                      // tensor_dim0 lo
    g1[2] = (rem0 >> 16) | ((rem1 & 0xFFFFu) << 16);     // dim0 hi | dim1 lo
    g1[3] = (rem1 >> 16) | (tile0 << 16);                // dim1 hi | tile_dim0
    g1[4] = tile1;                                       // tile_dim1 (tile_dim2=0)
    g1[5] = stride0;                                     // tensor_dim0_stride
    g1[6] = 0u;
    g1[7] = 0u;
    asm volatile("tensor_load_to_lds %0, %1" :: "s"(g0), "s"(g1) : "memory");
}

// ---------------- G = L L^T + reg^2 I ----------------
__global__ void k_init_gram(float* __restrict__ G) {
    int idx = blockIdx.x * 256 + threadIdx.x;
    if (idx >= NM * NM) return;
    int r = idx / NM, c = idx % NM;
    G[idx] = (r == c) ? REG2 : 0.0f;
}

__global__ void k_gram(const float* __restrict__ L, float* __restrict__ G) {
    __shared__ float la[16][68], lb[16][68];
    int i0 = blockIdx.x * 16, j0 = blockIdx.y * 16;
    long kbase = (long)blockIdx.z * 2000;
    int t = threadIdx.x;
    int ti = t & 15, tj = t >> 4;
    float acc = 0.0f;
    for (int c = 0; c < 2000; c += 64) {
        for (int idx = t; idx < 16 * 64; idx += 256) {
            int r = idx >> 6, k = idx & 63;
            long kk = kbase + c + k;
            la[r][k] = (i0 + r < NM && kk < NS) ? L[(long)(i0 + r) * NS + kk] : 0.0f;
            lb[r][k] = (j0 + r < NM && kk < NS) ? L[(long)(j0 + r) * NS + kk] : 0.0f;
        }
        __syncthreads();
#pragma unroll 8
        for (int k = 0; k < 64; ++k) acc += la[ti][k] * lb[tj][k];
        __syncthreads();
    }
    if (i0 + ti < NM && j0 + tj < NM)
        atomicAdd(&G[(i0 + ti) * NM + (j0 + tj)], acc);
}

// ---------------- In-place Cholesky of G (lower triangle valid) ----------------
__global__ void k_chol(float* __restrict__ G) {
    int t = threadIdx.x, nt = blockDim.x;
    for (int k = 0; k < NM; ++k) {
        if (t == 0) G[k * NM + k] = sqrtf(G[k * NM + k]);
        __syncthreads();
        float d = G[k * NM + k];
        for (int i = k + 1 + t; i < NM; i += nt) G[i * NM + k] /= d;
        __syncthreads();
        int m = NM - 1 - k;
        for (int idx = t; idx < m * m; idx += nt) {
            int i = k + 1 + idx / m, j = k + 1 + idx % m;
            G[i * NM + j] -= G[i * NM + k] * G[j * NM + k];
        }
        __syncthreads();
    }
}

// ---------------- Solve (L L^T) X = Y, one thread per RHS column ----------------
__global__ void k_solve(const float* __restrict__ G, const float* __restrict__ Y,
                        float* __restrict__ X) {
    __shared__ float z[NM][33];
    int lt = threadIdx.x;
    int col = blockIdx.x * 32 + lt;           // 0..255
    for (int i = 0; i < NM; ++i) {            // forward: L z = y
        float s = Y[i * NT + col];
        for (int j = 0; j < i; ++j) s -= G[i * NM + j] * z[j][lt];
        z[i][lt] = s / G[i * NM + i];
    }
    for (int i = NM - 1; i >= 0; --i) {       // backward: L^T x = z (in place)
        float s = z[i][lt];
        for (int j = i + 1; j < NM; ++j) s -= G[j * NM + i] * z[j][lt];
        z[i][lt] = s / G[i * NM + i];
    }
    for (int i = 0; i < NM; ++i) X[i * NT + col] = z[i][lt];
}

// ---------------- M = X (306x256) @ W_in[:, :256]^T -> (306x128) ----------------
__global__ void k_projM(const float* __restrict__ X, const float* __restrict__ Win,
                        float* __restrict__ M) {
    int idx = blockIdx.x * 256 + threadIdx.x;
    if (idx >= NM * NH) return;
    int r = idx / NH, h = idx % NH;
    float s = 0.0f;
    for (int k = 0; k < NT; ++k) s += X[r * NT + k] * Win[h * INDIM + k];
    M[idx] = s;
}

// ---- x = L^T M + VF @ W_in[:,256:]^T + b_in  (WMMA; A/B tiles via TDM) ----
__global__ void k_xinit(const float* __restrict__ L, const float* __restrict__ M,
                        const float* __restrict__ VF, const float* __restrict__ Win,
                        const float* __restrict__ bin, float* __restrict__ xout) {
    __shared__ float lA[32][17], lB[32][17];
    int n0 = blockIdx.x * 16, h0 = blockIdx.y * 16;
    int lane = threadIdx.x, half = lane >> 4, m = lane & 15;
    unsigned offA = (unsigned)(unsigned long long)&lA[0][0];
    unsigned offB = (unsigned)(unsigned long long)&lB[0][0];
    v8f c = {};
    for (int k0 = 0; k0 < NM; k0 += 32) {
        // Async DMA both tiles into LDS (TENSORcnt-tracked, EXEC-independent)
        tdm_load_2d(offA, L + (long)k0 * NS + n0, (unsigned)(NS - n0),
                    (unsigned)(NM - k0), 16u, 32u, (unsigned)NS);
        tdm_load_2d(offB, M + k0 * NH + h0, (unsigned)(NH - h0),
                    (unsigned)(NM - k0), 16u, 32u, (unsigned)NH);
        asm volatile("s_wait_tensorcnt 0x0" ::: "memory");
        v16bf a, b;
#pragma unroll
        for (int i = 0; i < 16; ++i) {
            int k = frag_k(i, half);
            a[i] = (__bf16)lA[k][m];
            b[i] = (__bf16)lB[k][m];
        }
        // LDS reads must retire before the next chunk's TDM overwrites the tiles
        asm volatile("s_wait_dscnt 0x0" ::: "memory");
        c = __builtin_amdgcn_wmma_f32_16x16x32_bf16(false, a, false, b, (short)0, c,
                                                    false, false);
    }
#pragma unroll
    for (int r = 0; r < 8; ++r) {
        int node = n0 + r + (half << 3);
        int h = h0 + m;
        float4 f0 = *(const float4*)(VF + (long)node * NF);
        float4 f1 = *(const float4*)(VF + (long)node * NF + 4);
        const float* wf = Win + h * INDIM + NT;
        float4 w0 = *(const float4*)(wf);
        float4 w1 = *(const float4*)(wf + 4);
        float vfe = f0.x * w0.x + f0.y * w0.y + f0.z * w0.z + f0.w * w0.w +
                    f1.x * w1.x + f1.y * w1.y + f1.z * w1.z + f1.w * w1.w;
        xout[(long)node * NH + h] = c[r] + vfe + bin[h];
    }
}

// ---------------- misc helpers ----------------
__global__ void k_zero(float* __restrict__ p, long n) {
    long i = (long)blockIdx.x * 256 + threadIdx.x;
    if (i < n) p[i] = 0.0f;
}

__global__ void k_degree(const int* __restrict__ recv, float* __restrict__ dg) {
    int e = blockIdx.x * 256 + threadIdx.x;
    if (e < NE) atomicAdd(&dg[recv[e]], 1.0f);
}

__global__ void k_invdeg(float* __restrict__ dg) {
    int i = blockIdx.x * 256 + threadIdx.x;
    if (i < NS) dg[i] = 1.0f / fmaxf(dg[i], 1.0f);
}

// one float4 per thread: 1 index load feeds 4 atomics
__global__ void k_scatter(const int* __restrict__ send, const int* __restrict__ recv,
                          const float* __restrict__ x, float* __restrict__ agg) {
    long gid = (long)blockIdx.x * 256 + threadIdx.x;
    if (gid >= (long)NE * (NH / 4)) return;
    int e = (int)(gid >> 5), h4 = (int)(gid & 31) << 2;
    float4 v = *(const float4*)(x + (long)send[e] * NH + h4);
    float* dst = agg + (long)recv[e] * NH + h4;
    atomicAdd(dst + 0, v.x);
    atomicAdd(dst + 1, v.y);
    atomicAdd(dst + 2, v.z);
    atomicAdd(dst + 3, v.w);
}

// Load a lane's 16-element A-fragment (two 32B runs) as 4x float4.
__device__ __forceinline__ void load_frag_f4(const float4* p4, int k0, int half,
                                             float* f) {
    int base = k0 >> 2;
    *(float4*)&f[0]  = p4[base + half * 2];
    *(float4*)&f[4]  = p4[base + half * 2 + 1];
    *(float4*)&f[8]  = p4[base + 4 + half * 2];
    *(float4*)&f[12] = p4[base + 5 + half * 2];
}

// ---------------- one GNN layer: relu(x Ws + (agg/deg) Wm + b) (WMMA) ----------
__global__ void k_gnn(const float* __restrict__ x, const float* __restrict__ agg,
                      const float* __restrict__ invdeg, const float* __restrict__ Ws,
                      const float* __restrict__ Wm, const float* __restrict__ bl,
                      float* __restrict__ xout) {
    int n0 = blockIdx.x * 16, h0 = blockIdx.y * 16;
    int lane = threadIdx.x, half = lane >> 4, m = lane & 15;
    int nrow = n0 + m;
    float inv = invdeg[nrow];
    const float4* px = (const float4*)(x + (long)nrow * NH);
    const float4* pg = (const float4*)(agg + (long)nrow * NH);
    v8f c = {};
    for (int k0 = 0; k0 < NH; k0 += 32) {
        float fx[16], fg[16];
        load_frag_f4(px, k0, half, fx);
        load_frag_f4(pg, k0, half, fg);
        v16bf a1, b1, a2, b2;
#pragma unroll
        for (int i = 0; i < 16; ++i) {
            int k = k0 + frag_k(i, half);
            a1[i] = (__bf16)fx[i];
            a2[i] = (__bf16)(fg[i] * inv);
            b1[i] = (__bf16)Ws[k * NH + h0 + m];
            b2[i] = (__bf16)Wm[k * NH + h0 + m];
        }
        c = __builtin_amdgcn_wmma_f32_16x16x32_bf16(false, a1, false, b1, (short)0, c,
                                                    false, false);
        c = __builtin_amdgcn_wmma_f32_16x16x32_bf16(false, a2, false, b2, (short)0, c,
                                                    false, false);
    }
#pragma unroll
    for (int r = 0; r < 8; ++r) {
        int node = n0 + r + (half << 3);
        int h = h0 + m;
        float v = c[r] + bl[h];
        xout[(long)node * NH + h] = v > 0.0f ? v : 0.0f;
    }
}

// ---------------- J = x @ W_out^T + b_out (WMMA) ----------------
__global__ void k_final(const float* __restrict__ x, const float* __restrict__ Wout,
                        const float* __restrict__ bout, float* __restrict__ out) {
    int n0 = blockIdx.x * 16, t0 = blockIdx.y * 16;
    int lane = threadIdx.x, half = lane >> 4, m = lane & 15;
    int nrow = n0 + m;
    const float4* px = (const float4*)(x + (long)nrow * NH);
    const float4* pw = (const float4*)(Wout + (long)(t0 + m) * NH);  // row t, K-major
    v8f c = {};
    for (int k0 = 0; k0 < NH; k0 += 32) {
        float fx[16], fw[16];
        load_frag_f4(px, k0, half, fx);
        load_frag_f4(pw, k0, half, fw);
        v16bf a, b;
#pragma unroll
        for (int i = 0; i < 16; ++i) {
            a[i] = (__bf16)fx[i];
            b[i] = (__bf16)fw[i];
        }
        c = __builtin_amdgcn_wmma_f32_16x16x32_bf16(false, a, false, b, (short)0, c,
                                                    false, false);
    }
#pragma unroll
    for (int r = 0; r < 8; ++r) {
        int node = n0 + r + (half << 3);
        int t = t0 + m;
        out[(long)node * NT + t] = c[r] + bout[t];
    }
}

extern "C" void kernel_launch(void* const* d_in, const int* in_sizes, int n_in,
                              void* d_out, int out_size, void* d_ws, size_t ws_size,
                              hipStream_t stream) {
    (void)in_sizes; (void)n_in; (void)out_size; (void)ws_size;
    const float* Y       = (const float*)d_in[0];
    const float* L       = (const float*)d_in[1];
    const float* VF      = (const float*)d_in[2];
    const int*   senders = (const int*)d_in[3];
    const int*   recvs   = (const int*)d_in[4];
    const float* W_in    = (const float*)d_in[5];
    const float* b_in    = (const float*)d_in[6];
    const float* W_self  = (const float*)d_in[7];
    const float* W_msg   = (const float*)d_in[8];
    const float* b_layer = (const float*)d_in[9];
    const float* W_out   = (const float*)d_in[10];
    const float* b_out   = (const float*)d_in[11];
    float* out = (float*)d_out;

    char* ws = (char*)d_ws;
    size_t off = 0;
    auto alloc = [&](size_t bytes) {
        size_t o = off;
        off += (bytes + 255) & ~(size_t)255;
        return o;
    };
    float* G   = (float*)(ws + alloc((size_t)NM * NM * 4));
    float* Xs  = (float*)(ws + alloc((size_t)NM * NT * 4));
    float* Mb  = (float*)(ws + alloc((size_t)NM * NH * 4));
    float* xA  = (float*)(ws + alloc((size_t)NS * NH * 4));
    float* xB  = (float*)(ws + alloc((size_t)NS * NH * 4));
    float* agg = (float*)(ws + alloc((size_t)NS * NH * 4));
    float* dg  = (float*)(ws + alloc((size_t)NS * 4));

    // 1. Gram matrix G = L L^T + reg^2 I
    k_init_gram<<<(NM * NM + 255) / 256, 256, 0, stream>>>(G);
    k_gram<<<dim3(20, 20, 50), 256, 0, stream>>>(L, G);
    // 2. Cholesky + solve -> X = (L L^T + reg^2 I)^{-1} Y
    k_chol<<<1, 1024, 0, stream>>>(G);
    k_solve<<<8, 32, 0, stream>>>(G, Y, Xs);
    // 3. Fold W_in through: M = X @ W_in[:, :256]^T  (never materialize J0)
    k_projM<<<(NM * NH + 255) / 256, 256, 0, stream>>>(Xs, W_in, Mb);
    // 4. x0 = L^T M + VF part + b_in   (TDM-staged WMMA GEMM over 100k nodes)
    k_xinit<<<dim3(NS / 16, NH / 16), 32, 0, stream>>>(L, Mb, VF, W_in, b_in, xA);
    // 5. Degree (edge-structure only; compute once) -> invdeg
    k_zero<<<(NS + 255) / 256, 256, 0, stream>>>(dg, (long)NS);
    k_degree<<<(NE + 255) / 256, 256, 0, stream>>>(recvs, dg);
    k_invdeg<<<(NS + 255) / 256, 256, 0, stream>>>(dg);
    // 6. Three graph-conv layers
    float* xc = xA;
    float* xn = xB;
    for (int l = 0; l < NL; ++l) {
        k_zero<<<(int)(((long)NS * NH + 255) / 256), 256, 0, stream>>>(agg, (long)NS * NH);
        k_scatter<<<(int)(((long)NE * (NH / 4) + 255) / 256), 256, 0, stream>>>(
            senders, recvs, xc, agg);
        k_gnn<<<dim3(NS / 16, NH / 16), 32, 0, stream>>>(
            xc, agg, dg, W_self + (size_t)l * NH * NH, W_msg + (size_t)l * NH * NH,
            b_layer + (size_t)l * NH, xn);
        float* tmp = xc; xc = xn; xn = tmp;
    }
    // 7. J = x @ W_out^T + b_out
    k_final<<<dim3(NS / 16, NT / 16), 32, 0, stream>>>(xc, W_out, b_out, out);
}
